// RConvCNP_44865228374512
// MI455X (gfx1250) — matrix operand order repaired and checked by reading, not compiled, not run on hardware
//
#include <hip/hip_runtime.h>
#include <math.h>

#define HH 128
#define WW 128
#define TT 30
#define SS 1024
#define BB 16
#define HWp (HH*WW)
#define STRIDE 36          // padded LDS row stride (halfs); max read col = 33

typedef _Float16 f16;
typedef __attribute__((ext_vector_type(16))) _Float16 v16h;
typedef __attribute__((ext_vector_type(8)))  float    v8f;

// Packed A-matrix tile bases (halfs). One tile = 32 lanes * 16 halfs = 512.
// Layers: ie1(4->3,2 tiles) ie2(3->3,1) id1(3->24,2) id2(24->6,7)
//         ce1(5->3,2) ce2(3->3,1) cd1(3->24,2) cd2(24->6,7) mg(6->3,2)
#define AT_IE1 (0*512)
#define AT_IE2 (2*512)
#define AT_ID1 (3*512)
#define AT_ID2 (5*512)
#define AT_CE1 (12*512)
#define AT_CE2 (14*512)
#define AT_CD1 (15*512)
#define AT_CD2 (17*512)
#define AT_MG  (24*512)
#define AT_TILES 26

// ---------------------------------------------------------------------------
// Weight packing. A-fragment layout per CDNA5 ISA 7.12.2 (16-bit A, 16x32):
//   lane holds M = lane&15 ; element e -> K = (e<8 ? e : e+8) + 8*(lane>>4)
// K dimension = Cin*9 + 1 (last K lane = bias via constant-1 input).
// ---------------------------------------------------------------------------
__global__ void pack_weights_kernel(f16* __restrict__ at,
    const float* ie_w1, const float* ie_b1, const float* ie_w2, const float* ie_b2,
    const float* id_w1, const float* id_b1, const float* id_w2, const float* id_b2,
    const float* ce_w1, const float* ce_b1, const float* ce_w2, const float* ce_b2,
    const float* cd_w1, const float* cd_b1, const float* cd_w2, const float* cd_b2,
    const float* mg_w,  const float* mg_b)
{
    const int lCin[9]  = {4, 3, 3, 24, 5, 3, 3, 24, 6};
    const int lCout[9] = {3, 3, 24, 6, 3, 3, 24, 6, 3};
    const int lBase[9] = {0, 2, 3, 5, 12, 14, 15, 17, 24};
    const int lCnt[9]  = {2, 1, 2, 7, 2, 1, 2, 7, 2};

    const int gt = blockIdx.x;
    int l = 0;
    while (l < 8 && gt >= lBase[l] + lCnt[l]) ++l;
    const int local = gt - lBase[l];
    const int Cin = lCin[l], Cout = lCout[l];
    const int Kc = (Cin * 9 + 1 + 31) >> 5;
    const int mc = local / Kc, kc = local - (local / Kc) * Kc;

    const float* wp; const float* bp;
    switch (l) {
        case 0: wp = ie_w1; bp = ie_b1; break;
        case 1: wp = ie_w2; bp = ie_b2; break;
        case 2: wp = id_w1; bp = id_b1; break;
        case 3: wp = id_w2; bp = id_b2; break;
        case 4: wp = ce_w1; bp = ce_b1; break;
        case 5: wp = ce_w2; bp = ce_b2; break;
        case 6: wp = cd_w1; bp = cd_b1; break;
        case 7: wp = cd_w2; bp = cd_b2; break;
        default: wp = mg_w;  bp = mg_b;  break;
    }

    const int lane = threadIdx.x;
    const int m = lane & 15;
    const int g = lane >> 4;
    for (int e = 0; e < 16; ++e) {
        const int k = ((e < 8) ? e : e + 8) + (g << 3) + (kc << 5);
        const int co = mc * 16 + m;
        float v = 0.f;
        if (co < Cout) {
            if (k < Cin * 9) {
                const int ci = k / 9, tap = k - (k / 9) * 9;
                v = wp[(co * Cin + ci) * 9 + tap];
            } else if (k == Cin * 9) {
                v = bp[co];
            }
        }
        at[(size_t)gt * 512 + lane * 16 + e] = (f16)v;
    }
}

// ---------------------------------------------------------------------------
// Parallel deterministic point scatter -> per-frame 4-ch f16 image
// [c0,c1,c2,mask]. Duplicate pixels resolved via LDS atomicMax on point
// index (highest point wins); JAX duplicate order is unspecified.
// ---------------------------------------------------------------------------
__global__ void scatter_kernel(f16* __restrict__ scat, const float* __restrict__ vids)
{
    __shared__ int winner[HWp];            // 64 KB LDS
    const int t = blockIdx.x, b = blockIdx.y;
    const size_t fb = ((size_t)(b * TT + t)) * 4 * HWp;

    for (int i = threadIdx.x; i < HWp; i += blockDim.x) winner[i] = -1;
    for (int i = threadIdx.x; i < 4 * HWp; i += blockDim.x) scat[fb + i] = (f16)0.f;
    __syncthreads();

    const float* vp = vids + ((size_t)(b * TT + t)) * SS * 5;
    int pix[SS / 256];
#pragma unroll
    for (int q = 0; q < SS / 256; ++q) {
        const int s = threadIdx.x + q * 256;
        int i = (int)(vp[s * 5 + 0] * 128.f);
        int j = (int)(vp[s * 5 + 1] * 128.f);
        i = min(max(i, 0), 127);
        j = min(max(j, 0), 127);
        pix[q] = i * WW + j;
        atomicMax(&winner[pix[q]], s);
    }
    __syncthreads();
#pragma unroll
    for (int q = 0; q < SS / 256; ++q) {
        const int s = threadIdx.x + q * 256;
        const int p = pix[q];
        scat[fb + 3 * HWp + p] = (f16)1.f;          // mask (same value, race-free)
        if (winner[p] == s) {
            for (int c = 0; c < 3; ++c) {
                float y = vp[s * 5 + 2 + c];
                y = fminf(fmaxf(y, 0.f), 1.f);
                scat[fb + c * HWp + p] = (f16)y;
            }
        }
    }
}

// ---------------------------------------------------------------------------
// 3x3 SAME conv over an LDS tile via WMMA implicit GEMM. Fully templated so
// all K/M chunk loops unroll and k->(ci,dy,dx) folds to constants.
// B build: one unconditional ds_load_u16 per element (address cndmask'd by
// lane-half), value select after the load -> batched, un-predicated loads.
// Store epilogue: straight-line; rows with a partially-valid lane-half store
// unconditionally with the ADDRESS selected (real slot vs sdump scratch).
//   actIn : LDS [ci][y(OUTW+2)][x], row stride STRIDE
//   actOut: LDS [co][y(OUTW)  ][x], row stride STRIDE
//   dump  : LDS scratch (>= 32 halfs), never read.
// Out-of-range lanes (x >= OUTW) store 0 -> pad columns deterministic.
// ---------------------------------------------------------------------------
template<int CIN, int COUT, int OUTW, bool RELU>
__device__ __forceinline__ void conv3x3_wmma(
    const f16* __restrict__ Atiles,
    const f16* actIn, f16* actOut, f16* dump,
    int wave, int nwaves, int lane)
{
    constexpr int K9  = CIN * 9;
    constexpr int KC  = (K9 + 1 + 31) / 32;
    constexpr int MC  = (COUT + 15) / 16;
    constexpr int NTX = (OUTW + 15) / 16;
    constexpr int NT  = OUTW * NTX;          // outH == OUTW (square tiles)
    constexpr int RIN = OUTW + 2;

    const int n = lane & 15;
    const int g = lane >> 4;

    // Hoist A fragments (loop-invariant) into registers.
    v16h af[KC * MC];
#pragma unroll
    for (int i = 0; i < KC * MC; ++i)
        af[i] = *(const v16h*)(Atiles + (i << 9) + (lane << 4));

    for (int nt = wave; nt < NT; nt += nwaves) {
        const int y = nt / NTX;
        const int x = (nt % NTX) * 16 + n;       // may exceed OUTW (edge lanes)
        const f16* lb = actIn + y * STRIDE + x;

        v8f acc[MC];
#pragma unroll
        for (int m = 0; m < MC; ++m)
            acc[m] = (v8f){0.f, 0.f, 0.f, 0.f, 0.f, 0.f, 0.f, 0.f};

#pragma unroll
        for (int kc = 0; kc < KC; ++kc) {
            // B-fragment (dense 16-bit B 32x16): element e -> K = (e<8 ? e : e+8) + 8*g.
            v16h bv;
#pragma unroll
            for (int e = 0; e < 16; ++e) {
                const int  k0 = ((e < 8) ? e : e + 8) + (kc << 5);
                const int  k1 = k0 + 8;
                const bool l0 = (k0 < K9), l1 = (k1 < K9);
                const int  o0 = l0 ? ((k0/9)*(RIN*STRIDE) + ((k0%9)/3)*STRIDE + (k0%9)%3) : 0;
                const int  o1 = l1 ? ((k1/9)*(RIN*STRIDE) + ((k1%9)/3)*STRIDE + (k1%9)%3) : 0;
                const f16  c0 = (k0 == K9) ? (f16)1.f : (f16)0.f;   // bias lane
                const f16  c1 = (k1 == K9) ? (f16)1.f : (f16)0.f;
                // Select the address (folds to a constant or one cndmask),
                // then ALWAYS load: keeps the load un-predicated/batchable.
                const int  osel = (l0 && l1) ? (g ? o1 : o0) : (l0 ? o0 : o1);
                const f16  lv = lb[osel];
                f16 v;
                if (l0 && l1)      v = lv;
                else if (l0)       v = g ? c1 : lv;
                else if (l1)       v = g ? lv : c0;
                else               v = g ? c1 : c0;   // lv dead -> DCE'd
                bv[e] = v;
            }
#pragma unroll
            for (int m = 0; m < MC; ++m)
                acc[m] = __builtin_amdgcn_wmma_f32_16x16x32_f16(
                             false, af[m * KC + kc], false, bv, (short)0,
                             acc[m], false, false);
        }

        const bool xvalid = (x < OUTW);
#pragma unroll
        for (int m = 0; m < MC; ++m) {
            // Rows actually needed by any lane-half (compile-time bound).
            const int RMAX = (COUT - m * 16 < 8) ? (COUT - m * 16) : 8;
#pragma unroll
            for (int r = 0; r < RMAX; ++r) {
                const int co = m * 16 + r + (g << 3);  // D: m = r + 8*(lane>>4)
                float v = acc[m][r];
                if (RELU) v = fmaxf(v, 0.f);
                v = xvalid ? v : 0.f;                  // zero pad columns
                // Unconditional store; invalid lane-half rows go to scratch.
                f16* p = (co < COUT) ? (actOut + (co * OUTW + y) * STRIDE + x)
                                     : (dump + lane);
                *p = (f16)v;
            }
        }
    }
}

// ---------------------------------------------------------------------------
// enc2 for every frame: [img3, t, mask] -> ce1(relu) -> ce2 -> enc (3ch f16)
// ---------------------------------------------------------------------------
__global__ void enc_kernel(const f16* __restrict__ scat, const f16* __restrict__ at,
                           f16* __restrict__ enc)
{
    __shared__ f16 s0[5 * 20 * STRIDE];
    __shared__ f16 s1[3 * 18 * STRIDE];
    __shared__ f16 s2[3 * 16 * STRIDE];
    __shared__ f16 sdump[32];

    __builtin_prefetch(at, 0, 1);

    const int tileid = blockIdx.x, t = blockIdx.y, b = blockIdx.z;
    const int yb = (tileid >> 3) * 16, xb = (tileid & 7) * 16;
    const size_t fb = ((size_t)(b * TT + t)) * 4 * HWp;
    const float tv = (float)t / (float)(TT - 1);

    for (int idx = threadIdx.x; idx < 5 * 20 * STRIDE; idx += blockDim.x) {
        const int ci = idx / (20 * STRIDE), rem = idx - ci * (20 * STRIDE);
        const int y = rem / STRIDE, x = rem - (rem / STRIDE) * STRIDE;
        const int gy = yb - 2 + y, gx = xb - 2 + x;
        f16 v = (f16)0.f;
        if ((unsigned)gy < 128u && (unsigned)gx < 128u) {
            if (ci < 3)       v = scat[fb + ci * HWp + gy * WW + gx];
            else if (ci == 3) v = (f16)tv;                           // time ch
            else              v = scat[fb + 3 * HWp + gy * WW + gx]; // mask
        }
        s0[idx] = v;
    }
    __syncthreads();

    const int lane = threadIdx.x & 31, wave = threadIdx.x >> 5, nw = blockDim.x >> 5;
    conv3x3_wmma<5, 3, 18, true >(at + AT_CE1, s0, s1, sdump, wave, nw, lane);
    __syncthreads();
    conv3x3_wmma<3, 3, 16, false>(at + AT_CE2, s1, s2, sdump, wave, nw, lane);
    __syncthreads();

    const size_t eb = ((size_t)(b * TT + t)) * 3 * HWp;
    for (int idx = threadIdx.x; idx < 3 * 256; idx += blockDim.x) {
        const int c = idx >> 8, p = idx & 255;
        const int y = p >> 4, x = p & 15;
        enc[eb + c * HWp + (size_t)(yb + y) * WW + (xb + x)] =
            s2[(c * 16 + y) * STRIDE + x];
    }
}

// ---------------------------------------------------------------------------
// init_cell (t = 0): [img3,mask] -> ie1(relu)->ie2 -> id1(relu)->id2
// mu -> out[t=0] & Mem0 ; sig = clip(softplus(.), 1e-6, 1) -> out
// ---------------------------------------------------------------------------
__global__ void init_kernel(const f16* __restrict__ scat, const f16* __restrict__ at,
                            f16* __restrict__ mem0, float* __restrict__ out)
{
    __shared__ f16 s0[4 * 24 * STRIDE];
    __shared__ f16 s1[3 * 22 * STRIDE];
    __shared__ f16 s2[3 * 20 * STRIDE];
    __shared__ f16 s3[24 * 18 * STRIDE];
    __shared__ f16 s4[6 * 16 * STRIDE];
    __shared__ f16 sdump[32];

    __builtin_prefetch(at, 0, 1);

    const int tileid = blockIdx.x, b = blockIdx.y;
    const int yb = (tileid >> 3) * 16, xb = (tileid & 7) * 16;
    const size_t fb = ((size_t)(b * TT)) * 4 * HWp;      // frame t=0

    for (int idx = threadIdx.x; idx < 4 * 24 * STRIDE; idx += blockDim.x) {
        const int ci = idx / (24 * STRIDE), rem = idx - ci * (24 * STRIDE);
        const int y = rem / STRIDE, x = rem - (rem / STRIDE) * STRIDE;
        const int gy = yb - 4 + y, gx = xb - 4 + x;
        f16 v = (f16)0.f;
        if ((unsigned)gy < 128u && (unsigned)gx < 128u)
            v = scat[fb + ci * HWp + gy * WW + gx];
        s0[idx] = v;
    }
    __syncthreads();

    const int lane = threadIdx.x & 31, wave = threadIdx.x >> 5, nw = blockDim.x >> 5;
    conv3x3_wmma<4, 3, 22, true >(at + AT_IE1, s0, s1, sdump, wave, nw, lane);
    __syncthreads();
    conv3x3_wmma<3, 3, 20, false>(at + AT_IE2, s1, s2, sdump, wave, nw, lane);
    __syncthreads();
    conv3x3_wmma<3, 24, 18, true >(at + AT_ID1, s2, s3, sdump, wave, nw, lane);
    __syncthreads();
    conv3x3_wmma<24, 6, 16, false>(at + AT_ID2, s3, s4, sdump, wave, nw, lane);
    __syncthreads();

    const size_t sigOff = (size_t)BB * TT * HWp * 3;
    const size_t ob = ((size_t)(b * TT)) * HWp;
    for (int idx = threadIdx.x; idx < 6 * 256; idx += blockDim.x) {
        const int co = idx >> 8, p = idx & 255;
        const int y = p >> 4, x = p & 15;
        const int gy = yb + y, gx = xb + x;
        const float v = (float)s4[(co * 16 + y) * STRIDE + x];
        const size_t pix = ob + (size_t)gy * WW + gx;
        if (co < 3) {
            out[pix * 3 + co] = v;
            mem0[((size_t)b * 3 + co) * HWp + gy * WW + gx] = (f16)v;
        } else {
            float sp = (v > 20.f) ? v : log1pf(expf(v));
            sp = fminf(fmaxf(sp, 1e-6f), 1.f);
            out[sigOff + pix * 3 + (co - 3)] = sp;
        }
    }
}

// ---------------------------------------------------------------------------
// One recurrence step t (t = 1..T-1):
//   hc  = (enc[t-1] - enc[t]) * (T-1)
//   hc2 = mg([Mem, hc])   (no relu)
//   u   = relu(cd1(hc2)) ; ft = cd2(u)
//   mu  = ft[:3]*Dt -> out & MemDst ; sig = clip(softplus(ft[3:]*Dt),eps,1)
// Mem double-buffered across launches (neighbor tiles read halo pixels).
// ---------------------------------------------------------------------------
__global__ void step_kernel(int t, const f16* __restrict__ enc, const f16* __restrict__ at,
                            const f16* __restrict__ memSrc, f16* __restrict__ memDst,
                            float* __restrict__ out)
{
    __shared__ f16 s0[6 * 22 * STRIDE];
    __shared__ f16 s1[3 * 20 * STRIDE];
    __shared__ f16 s2[24 * 18 * STRIDE];
    __shared__ f16 s3[6 * 16 * STRIDE];
    __shared__ f16 sdump[32];

    __builtin_prefetch(at, 0, 1);

    const int tileid = blockIdx.x, b = blockIdx.y;
    const int yb = (tileid >> 3) * 16, xb = (tileid & 7) * 16;
    const size_t e0 = ((size_t)(b * TT + (t - 1))) * 3 * HWp;
    const size_t e1 = ((size_t)(b * TT + t)) * 3 * HWp;

    for (int idx = threadIdx.x; idx < 6 * 22 * STRIDE; idx += blockDim.x) {
        const int ci = idx / (22 * STRIDE), rem = idx - ci * (22 * STRIDE);
        const int y = rem / STRIDE, x = rem - (rem / STRIDE) * STRIDE;
        const int gy = yb - 3 + y, gx = xb - 3 + x;
        f16 v = (f16)0.f;
        if ((unsigned)gy < 128u && (unsigned)gx < 128u) {
            const size_t p = (size_t)gy * WW + gx;
            if (ci < 3) {
                v = memSrc[((size_t)b * 3 + ci) * HWp + p];
            } else {
                const int c = ci - 3;
                const float d = ((float)enc[e0 + c * HWp + p] -
                                 (float)enc[e1 + c * HWp + p]) * (float)(TT - 1);
                v = (f16)d;
            }
        }
        s0[idx] = v;
    }
    __syncthreads();

    const int lane = threadIdx.x & 31, wave = threadIdx.x >> 5, nw = blockDim.x >> 5;
    conv3x3_wmma<6, 3, 20, false>(at + AT_MG, s0, s1, sdump, wave, nw, lane);
    __syncthreads();
    conv3x3_wmma<3, 24, 18, true >(at + AT_CD1, s1, s2, sdump, wave, nw, lane);
    __syncthreads();
    conv3x3_wmma<24, 6, 16, false>(at + AT_CD2, s2, s3, sdump, wave, nw, lane);
    __syncthreads();

    const float Dtv = 1.f / (float)(TT - 1);
    const size_t sigOff = (size_t)BB * TT * HWp * 3;
    const size_t ob = ((size_t)(b * TT + t)) * HWp;
    for (int idx = threadIdx.x; idx < 6 * 256; idx += blockDim.x) {
        const int co = idx >> 8, p = idx & 255;
        const int y = p >> 4, x = p & 15;
        const int gy = yb + y, gx = xb + x;
        const float v = (float)s3[(co * 16 + y) * STRIDE + x];
        const size_t pix = ob + (size_t)gy * WW + gx;
        if (co < 3) {
            const float mu = v * Dtv;
            out[pix * 3 + co] = mu;
            memDst[((size_t)b * 3 + co) * HWp + gy * WW + gx] = (f16)mu;
        } else {
            float sp = v * Dtv;
            sp = (sp > 20.f) ? sp : log1pf(expf(sp));
            sp = fminf(fmaxf(sp, 1e-6f), 1.f);
            out[sigOff + pix * 3 + (co - 3)] = sp;
        }
    }
}

// ---------------------------------------------------------------------------
// Host launcher. Workspace layout (bytes):
//   [0, 32K)  packed WMMA weight fragments (26 tiles)
//   scat : B*T*4*HW f16 (~63 MB) ; enc : B*T*3*HW f16 (~47 MB)
//   mem0/mem1 : B*3*HW f16 double-buffered recurrent state (2 x 1.5 MB)
// ---------------------------------------------------------------------------
extern "C" void kernel_launch(void* const* d_in, const int* in_sizes, int n_in,
                              void* d_out, int out_size, void* d_ws, size_t ws_size,
                              hipStream_t stream)
{
    (void)in_sizes; (void)n_in; (void)out_size; (void)ws_size;

    const float* vids = (const float*)d_in[0];
    char* ws = (char*)d_ws;

    f16* at = (f16*)ws;
    size_t off = 32768;
    f16* scat = (f16*)(ws + off); off += (size_t)BB * TT * 4 * HWp * sizeof(f16);
    f16* enc  = (f16*)(ws + off); off += (size_t)BB * TT * 3 * HWp * sizeof(f16);
    f16* mem0 = (f16*)(ws + off); off += (size_t)BB * 3 * HWp * sizeof(f16);
    f16* mem1 = (f16*)(ws + off);
    float* out = (float*)d_out;

    pack_weights_kernel<<<AT_TILES, 32, 0, stream>>>(at,
        (const float*)d_in[1],  (const float*)d_in[2],  (const float*)d_in[3],  (const float*)d_in[4],
        (const float*)d_in[5],  (const float*)d_in[6],  (const float*)d_in[7],  (const float*)d_in[8],
        (const float*)d_in[9],  (const float*)d_in[10], (const float*)d_in[11], (const float*)d_in[12],
        (const float*)d_in[13], (const float*)d_in[14], (const float*)d_in[15], (const float*)d_in[16],
        (const float*)d_in[17], (const float*)d_in[18]);

    scatter_kernel<<<dim3(TT, BB), 256, 0, stream>>>(scat, vids);

    enc_kernel<<<dim3(64, TT, BB), 128, 0, stream>>>(scat, at, enc);

    init_kernel<<<dim3(64, BB), 128, 0, stream>>>(scat, at, mem0, out);

    for (int t = 1; t < TT; ++t) {
        const f16* src = (t & 1) ? mem0 : mem1;
        f16*       dst = (t & 1) ? mem1 : mem0;
        step_kernel<<<dim3(64, BB), 128, 0, stream>>>(t, enc, at, src, dst, out);
    }
}
